// TransformerDecoder_10127532884297
// MI455X (gfx1250) — matrix operand order repaired and checked by reading, not compiled, not run on hardware
//
#include <hip/hip_runtime.h>
#include <hip/hip_bf16.h>
#include <math.h>

// ---------------------------------------------------------------------------
// Transformer decoder forward for MI455X (gfx1250, wave32, WMMA).
// GEMMs use V_WMMA_F32_16X16X32_BF16 (fp32 data RNE-rounded to bf16, f32 acc).
// ---------------------------------------------------------------------------

#define LNUM 2
#define HNUM 12
#define DKV  64
#define DMOD 768
#define DHID 3072
#define VOC  32000
#define SEQ  1024
#define BAT  2
#define ROWS (BAT * SEQ)          // 2048 token rows

typedef __bf16 bf16_t;
typedef bf16_t v16bf __attribute__((ext_vector_type(16)));
typedef float  v8f   __attribute__((ext_vector_type(8)));

union Frag16 { unsigned u[8]; v16bf v; };
union FragAcc { float f[8]; v8f v; };

__device__ __forceinline__ unsigned pack_bf16x2(float a, float b) {
    unsigned ua = __float_as_uint(a);
    unsigned ub = __float_as_uint(b);
    unsigned ra = (ua + 0x7FFFu + ((ua >> 16) & 1u)) >> 16;   // RNE to bf16
    unsigned rb = (ub + 0x7FFFu + ((ub >> 16) & 1u)) >> 16;
    return (rb << 16) | (ra & 0xFFFFu);
}

// ---------------------------------------------------------------------------
// Embedding: h[b,s,:] = emb[x[b,s]] + pos[s] * (ignore[b,s]==0)
// ---------------------------------------------------------------------------
__global__ void embed_kernel(const int* __restrict__ x, const int* __restrict__ ign,
                             const float* __restrict__ emb, const float* __restrict__ pos,
                             float* __restrict__ h) {
    int idx = blockIdx.x * blockDim.x + threadIdx.x;
    if (idx >= ROWS * DMOD) return;
    int r = idx / DMOD, d = idx - r * DMOD;
    int tok = x[r];
    float p = (ign[r] == 0) ? pos[(r % SEQ) * DMOD + d] : 0.0f;
    h[idx] = emb[(size_t)tok * DMOD + d] + p;
}

// ---------------------------------------------------------------------------
// bf16 WMMA GEMM:  C[M,N] = act(A[M,K] * B + bias)
//   A: fp32 row-major, lda.
//   B: fp32, element (d, j) at d*ldb + (j>>6)*hstride + (j&63).
//      plain row-major (N cols): ldb=N, hstride=64
//      per-head (H,D,64):        ldb=64, hstride=D*64
//   act: 0=none, 1=exact GELU.
//
// Block tile 256(M) x 64(N), K-step 32, 8 waves. Each wave computes 32x64:
// 2 A-fragments x 4 B-fragments -> 8 WMMAs per K-step per wave.
// LDS tiles are stored FRAGMENT-MAJOR so every lane's fragment is two
// contiguous 16-byte chunks -> ds_load_b128, no v_mov shuffling.
// ---------------------------------------------------------------------------
#define TBM 256
#define TBN 64
#define TBK 32
#define AWORDS (TBM * (TBK / 2))   // 4096 words, [row][kpair]
#define BWORDS (TBN * (TBK / 2))   // 1024 words, [col][kpair]

__global__ __launch_bounds__(256) void gemm_bf16_wmma(
    const float* __restrict__ A, const float* __restrict__ B,
    const float* __restrict__ bias, float* __restrict__ C,
    int K, int lda, int ldb, long hstride, int ldc, int act)
{
    __shared__ unsigned As[AWORDS];   // [row][kpair]  256 x 16 words
    __shared__ unsigned Bs[BWORDS];   // [col][kpair]   64 x 16 words

    const int tid   = threadIdx.x;
    const int wv    = tid >> 5;
    const int lane  = tid & 31;
    const int lhalf = (lane >> 4) & 1;
    const int m16   = lane & 15;
    const int tileM = blockIdx.y * TBM;
    const int tileN = blockIdx.x * TBN;

    FragAcc acc[2][4];
#pragma unroll
    for (int sm = 0; sm < 2; ++sm)
#pragma unroll
        for (int ct = 0; ct < 4; ++ct)
#pragma unroll
            for (int r = 0; r < 8; ++r) acc[sm][ct].f[r] = 0.0f;

    for (int k0 = 0; k0 < K; k0 += TBK) {
        // ---- Stage A tile (fp32 -> packed bf16x2 along K), [row][kpair] ----
#pragma unroll
        for (int i = tid; i < AWORDS; i += 256) {
            int row = i >> 4, c = i & 15;
            const float* g = A + (size_t)(tileM + row) * lda + k0 + 2 * c;
            As[i] = pack_bf16x2(g[0], g[1]);
            if (k0 + TBK < K)
                __builtin_prefetch(g + TBK, 0, 0);   // global_prefetch_b8: next K tile
        }
        // ---- Stage B tile, fragment-major [col][kpair] ----
#pragma unroll
        for (int i = tid; i < BWORDS; i += 256) {
            int n = i >> 4, p = i & 15;
            int j = tileN + n;
            size_t off = (size_t)(k0 + 2 * p) * ldb + (size_t)(j >> 6) * hstride + (j & 63);
            Bs[i] = pack_bf16x2(B[off], B[off + ldb]);
            if (k0 + TBK < K)
                __builtin_prefetch(&B[off + (size_t)TBK * ldb], 0, 0);
        }
        __syncthreads();

        // A fragments (16x32 bf16 each): VGPR v -> pair (v&3) + lhalf*4 + (v>=4)*8,
        // i.e. two contiguous 16B chunks at pair offsets {lhalf*4, lhalf*4+8}.
        Frag16 a[2];
#pragma unroll
        for (int sm = 0; sm < 2; ++sm) {
            int base = (wv * 32 + sm * 16 + m16) * 16 + lhalf * 4;
#pragma unroll
            for (int vv = 0; vv < 4; ++vv) {
                a[sm].u[vv]     = As[base + vv];
                a[sm].u[vv + 4] = As[base + 8 + vv];
            }
        }
#pragma unroll
        for (int ct = 0; ct < 4; ++ct) {
            // B fragment (32x16 bf16): VGPR v -> pair (v&3) + lhalf*4 + (v>=4)*8 (same as A)
            Frag16 bfr;
            int base = (ct * 16 + m16) * 16 + lhalf * 4;
#pragma unroll
            for (int vv = 0; vv < 4; ++vv) {
                bfr.u[vv]     = Bs[base + vv];
                bfr.u[vv + 4] = Bs[base + 8 + vv];
            }
#pragma unroll
            for (int sm = 0; sm < 2; ++sm) {
                acc[sm][ct].v = __builtin_amdgcn_wmma_f32_16x16x32_bf16(
                    false, a[sm].v, false, bfr.v, (short)0, acc[sm][ct].v, false, false);
            }
        }
        __syncthreads();
    }

    // Epilogue: C layout VGPR r -> M = r + lhalf*8, N = m16
#pragma unroll
    for (int ct = 0; ct < 4; ++ct) {
        int col = tileN + ct * 16 + m16;
        float bval = bias ? bias[col] : 0.0f;
#pragma unroll
        for (int sm = 0; sm < 2; ++sm) {
#pragma unroll
            for (int r = 0; r < 8; ++r) {
                int row = tileM + wv * 32 + sm * 16 + r + lhalf * 8;
                float val = acc[sm][ct].f[r] + bval;
                if (act == 1) val = 0.5f * val * (1.0f + erff(val * 0.70710678118654752f));
                C[(size_t)row * ldc + col] = val;
            }
        }
    }
}

// ---------------------------------------------------------------------------
// Fused causal masked attention, online softmax. One wave32 per (b,h,i).
// q/k/v layout: [B,S, H*64] (head-major columns, matches torch cat order).
// mask: attend iff j<=i AND (ignore[j]==0 OR j==i).
// ---------------------------------------------------------------------------
__global__ __launch_bounds__(256) void attn_kernel(
    const float* __restrict__ q, const float* __restrict__ k,
    const float* __restrict__ v, const int* __restrict__ ign,
    float* __restrict__ sa)
{
    int gwave = (blockIdx.x * blockDim.x + threadIdx.x) >> 5;
    int lane  = threadIdx.x & 31;
    if (gwave >= BAT * HNUM * SEQ) return;
    int b   = gwave / (HNUM * SEQ);
    int rem = gwave - b * (HNUM * SEQ);
    int hh  = rem / SEQ;
    int i   = rem - hh * SEQ;

    const float* qp = q + ((size_t)(b * SEQ + i)) * DMOD + hh * DKV;
    float q0 = qp[2 * lane], q1 = qp[2 * lane + 1];

    float m = -INFINITY, lsum = 0.0f, o0 = 0.0f, o1 = 0.0f;
    const int* igb = ign + b * SEQ;

    for (int j = 0; j <= i; ++j) {
        if (igb[j] != 0 && j != i) continue;           // masked position
        const float* kp = k + ((size_t)(b * SEQ + j)) * DMOD + hh * DKV;
        float part = q0 * kp[2 * lane] + q1 * kp[2 * lane + 1];
#pragma unroll
        for (int off = 16; off > 0; off >>= 1)
            part += __shfl_xor(part, off, 32);          // wave32 reduction
        float s  = part * 0.125f;                       // 1/sqrt(64)
        float nm = fmaxf(m, s);
        float f  = __expf(m - nm);                      // 0 on first hit (m=-inf)
        float p  = __expf(s - nm);
        lsum = lsum * f + p;
        const float* vp = v + ((size_t)(b * SEQ + j)) * DMOD + hh * DKV;
        o0 = o0 * f + p * vp[2 * lane];
        o1 = o1 * f + p * vp[2 * lane + 1];
        m = nm;
    }
    float inv = 1.0f / lsum;
    float* op = sa + ((size_t)(b * SEQ + i)) * DMOD + hh * DKV;
    op[2 * lane]     = o0 * inv;
    op[2 * lane + 1] = o1 * inv;
}

// ---------------------------------------------------------------------------
// out = LayerNorm(a + b): mean over last dim, biased std, no eps, no affine.
// One block (256 threads) per row of 768.
// ---------------------------------------------------------------------------
__global__ __launch_bounds__(256) void ln_add_kernel(
    const float* __restrict__ a, const float* __restrict__ b, float* __restrict__ out)
{
    __shared__ float red[256];
    int row = blockIdx.x, tid = threadIdx.x;
    const float* ar = a + (size_t)row * DMOD;
    const float* br = b + (size_t)row * DMOD;

    float s = 0.0f;
    for (int d = tid; d < DMOD; d += 256) s += ar[d] + br[d];
    red[tid] = s; __syncthreads();
    for (int off = 128; off > 0; off >>= 1) {
        if (tid < off) red[tid] += red[tid + off];
        __syncthreads();
    }
    float mu = red[0] * (1.0f / DMOD);
    __syncthreads();

    float s2 = 0.0f;
    for (int d = tid; d < DMOD; d += 256) {
        float t = ar[d] + br[d] - mu; s2 += t * t;
    }
    red[tid] = s2; __syncthreads();
    for (int off = 128; off > 0; off >>= 1) {
        if (tid < off) red[tid] += red[tid + off];
        __syncthreads();
    }
    float inv = 1.0f / sqrtf(red[0] * (1.0f / DMOD));
    for (int d = tid; d < DMOD; d += 256)
        out[(size_t)row * DMOD + d] = (ar[d] + br[d] - mu) * inv;
}

// ---------------------------------------------------------------------------
extern "C" void kernel_launch(void* const* d_in, const int* in_sizes, int n_in,
                              void* d_out, int out_size, void* d_ws, size_t ws_size,
                              hipStream_t stream) {
    const int*   x    = (const int*)d_in[0];
    const int*   ign  = (const int*)d_in[1];
    const float* emb  = (const float*)d_in[2];
    const float* pos  = (const float*)d_in[3];
    const float* Wq   = (const float*)d_in[4];
    const float* bq   = (const float*)d_in[5];
    const float* Wk   = (const float*)d_in[6];
    const float* bk   = (const float*)d_in[7];
    const float* Wv   = (const float*)d_in[8];
    const float* bv   = (const float*)d_in[9];
    const float* Wo   = (const float*)d_in[10];
    const float* bo   = (const float*)d_in[11];
    const float* W1   = (const float*)d_in[12];
    const float* b1   = (const float*)d_in[13];
    const float* W2   = (const float*)d_in[14];
    const float* b2   = (const float*)d_in[15];
    const float* Wout = (const float*)d_in[16];
    const float* bout = (const float*)d_in[17];
    float* out = (float*)d_out;

    // Workspace carve-up (floats). Needs 12 * 2048*768 * 4B = ~75.5 MB.
    const size_t RD = (size_t)ROWS * DMOD;
    float* h    = (float*)d_ws;
    float* qb   = h    + RD;
    float* kb   = qb   + RD;
    float* vb   = kb   + RD;
    float* sab  = vb   + RD;
    float* atb  = sab  + RD;
    float* zb   = atb  + RD;
    float* fb   = zb   + RD;
    float* t1   = fb   + RD;      // 2048 x 3072 (4*RD floats)

    // 1) Embedding
    embed_kernel<<<(ROWS * DMOD + 255) / 256, 256, 0, stream>>>(x, ign, emb, pos, h);

    const dim3 g768(DMOD / TBN, ROWS / TBM);     // 12 x 8
    const dim3 g3072(DHID / TBN, ROWS / TBM);    // 48 x 8
    const long headStride = (long)DMOD * DKV;    // per-head weight stride (H,D,64)

    for (int l = 0; l < LNUM; ++l) {
        const float* Wql = Wq + (size_t)l * HNUM * DMOD * DKV;
        const float* Wkl = Wk + (size_t)l * HNUM * DMOD * DKV;
        const float* Wvl = Wv + (size_t)l * HNUM * DMOD * DKV;
        const float* Wol = Wo + (size_t)l * DMOD * DMOD;
        const float* W1l = W1 + (size_t)l * DMOD * DHID;
        const float* W2l = W2 + (size_t)l * DHID * DMOD;

        // QKV projections (per-head weights addressed via hstride)
        gemm_bf16_wmma<<<g768, 256, 0, stream>>>(h, Wql, bq + l * DMOD, qb,
                                                 DMOD, DMOD, DKV, headStride, DMOD, 0);
        gemm_bf16_wmma<<<g768, 256, 0, stream>>>(h, Wkl, bk + l * DMOD, kb,
                                                 DMOD, DMOD, DKV, headStride, DMOD, 0);
        gemm_bf16_wmma<<<g768, 256, 0, stream>>>(h, Wvl, bv + l * DMOD, vb,
                                                 DMOD, DMOD, DKV, headStride, DMOD, 0);

        // Attention: 24576 wave32's, 8 per block
        attn_kernel<<<(BAT * HNUM * SEQ) / 8, 256, 0, stream>>>(qb, kb, vb, ign, sab);

        // Output projection
        gemm_bf16_wmma<<<g768, 256, 0, stream>>>(sab, Wol, bo + l * DMOD, atb,
                                                 DMOD, DMOD, DMOD, 64, DMOD, 0);

        // z = LN(h + attn_out)
        ln_add_kernel<<<ROWS, 256, 0, stream>>>(h, atb, zb);

        // FFN: t1 = z@W1 + b1 ; f = gelu(t1@W2 + b2)
        gemm_bf16_wmma<<<g3072, 256, 0, stream>>>(zb, W1l, b1 + l * DHID, t1,
                                                  DMOD, DMOD, DHID, 64, DHID, 0);
        gemm_bf16_wmma<<<g768, 256, 0, stream>>>(t1, W2l, b2 + l * DMOD, fb,
                                                 DHID, DHID, DMOD, 64, DMOD, 1);

        // h = LN(z + f)
        ln_add_kernel<<<ROWS, 256, 0, stream>>>(zb, fb, h);
    }

    // Final vocab projection (the compute-dominant GEMM: 100.7 GF)
    const dim3 gout(VOC / TBN, ROWS / TBM);      // 500 x 8
    gemm_bf16_wmma<<<gout, 256, 0, stream>>>(h, Wout, bout, out,
                                             DMOD, DMOD, VOC, 64, VOC, 0);
}
